// CognitiveKernel_420906795400
// MI455X (gfx1250) — compile-verified, matrix operand order checked by reading, not run on hardware
//
#include <hip/hip_runtime.h>
#include <hip/hip_bf16.h>

#define B_SZ   32768
#define COORD  64
#define DG     2048
#define CA3    256
#define CA1    256
#define K_ACT  81
#define STEPS  5
#define GAIN   0.7f
#define LN_EPS 1e-5f
#define COS_EPSF 1e-8f

typedef _Float16 half16 __attribute__((ext_vector_type(16)));
typedef _Float16 half8  __attribute__((ext_vector_type(8)));
typedef float    float8 __attribute__((ext_vector_type(8)));

static __device__ __forceinline__ float8 wmma_f16(half16 a, half16 b, float8 c) {
    return __builtin_amdgcn_wmma_f32_16x16x32_f16(false, a, false, b, (short)0, c, false, false);
}
static __device__ __forceinline__ half16 cat16(half8 lo, half8 hi) {
    return __builtin_shufflevector(lo, hi, 0,1,2,3,4,5,6,7,8,9,10,11,12,13,14,15);
}

// A-fragment (16xK tile, 32-wide K chunk at kbase) from f16 LDS, row stride ld.
// 16-B aligned vector loads -> ds_load_b128.
static __device__ __forceinline__ half16 ldsA_h(const _Float16* s, int ld, int kbase, int lane) {
    int m   = lane & 15;
    int klo = (lane & 16) ? 8 : 0;
    const half8* p = (const half8*)(s + m * ld + kbase + klo);
    return cat16(p[0], p[2]);   // halves K=klo..klo+7 and K=16+klo..16+klo+7
}
// A-fragment from f32 LDS (converted to f16) -- only used for the tiny ec tile.
static __device__ __forceinline__ half16 ldsA_f(const float* s, int ld, int kbase, int lane) {
    int m   = lane & 15;
    int klo = (lane & 16) ? 8 : 0;
    const float* p = s + m * ld + kbase + klo;
    half16 a;
#pragma unroll
    for (int i = 0; i < 8; ++i) { a[i] = (_Float16)p[i]; a[i + 8] = (_Float16)p[16 + i]; }
    return a;
}
// B-fragment (Kx16 tile) from a pre-converted row-major [N][K] f16 weight.
static __device__ __forceinline__ half16 gB_h(const _Float16* __restrict__ W, int ld, int nb, int kbase, int lane) {
    int n   = nb + (lane & 15);
    int klo = (lane & 16) ? 8 : 0;
    const half8* p = (const half8*)(W + (size_t)n * ld + kbase + klo);
    return cat16(p[0], p[2]);   // two global_load_b128
}

static __device__ __forceinline__ float halfWaveSum(float v) {
#pragma unroll
    for (int m = 1; m <= 8; m <<= 1) v += __shfl_xor(v, m, 32);
    return v;
}
static __device__ __forceinline__ float waveSum(float v) {
#pragma unroll
    for (int m = 1; m <= 16; m <<= 1) v += __shfl_xor(v, m, 32);
    return v;
}
static __device__ __forceinline__ float waveMax(float v) {
#pragma unroll
    for (int m = 1; m <= 16; m <<= 1) v = fmaxf(v, __shfl_xor(v, m, 32));
    return v;
}
static __device__ __forceinline__ float waveMin(float v) {
#pragma unroll
    for (int m = 1; m <= 16; m <<= 1) v = fminf(v, __shfl_xor(v, m, 32));
    return v;
}

// ---------------- prep kernels ----------------------------------------------
__global__ void __launch_bounds__(256) hippo_colsum(const float* __restrict__ coords,
                                                    float* __restrict__ sums) {
    __shared__ float part[256];
    int tid = threadIdx.x;
    int col = tid & 63;
    int sub = tid >> 6;
    int r0  = blockIdx.x * 128;
    float s = 0.0f;
    for (int r = r0 + sub; r < r0 + 128; r += 4)
        s += coords[(size_t)r * COORD + col];
    part[tid] = s;
    __syncthreads();
    if (tid < 64)
        atomicAdd(&sums[col], part[tid] + part[tid + 64] + part[tid + 128] + part[tid + 192]);
}

// f32 -> f16 weight copy (one-time per launch; removes cvts from hot GEMM loops)
__global__ void __launch_bounds__(256) cvt_f16(const float* __restrict__ src,
                                               _Float16* __restrict__ dst, int n) {
    int i = blockIdx.x * 256 + threadIdx.x;
    if (i < n) dst[i] = (_Float16)src[i];
}

// W_mossy [CA3][DG] -> W_mossyT [DG][CA3] so the sparse gather is coalesced.
__global__ void __launch_bounds__(256) mossy_T(const float* __restrict__ W,
                                               float* __restrict__ WT) {
    int e = blockIdx.x * 256 + threadIdx.x;   // e over DG*CA3
    int n = e >> 8, c = e & 255;
    WT[(size_t)n * CA3 + c] = W[(size_t)c * DG + n];
}

// ---------------- fused pipeline, 16 rows/block ------------------------------
__global__ void __launch_bounds__(256) hippo_fused(
    const float* __restrict__ coords, const float* __restrict__ ec_gamma,
    const float* __restrict__ ec_beta,
    const _Float16* __restrict__ Wpp_h, const float* __restrict__ b_pp,
    const float* __restrict__ dg_gamma, const float* __restrict__ dg_beta,
    const float* __restrict__ W_mossyT,
    const _Float16* __restrict__ Wrec_h, const _Float16* __restrict__ Wsc_h,
    const float* __restrict__ b_sc,
    const _Float16* __restrict__ Wta_h, const float* __restrict__ b_ta,
    const _Float16* __restrict__ Wcs_h, const _Float16* __restrict__ Wcd_h,
    const _Float16* __restrict__ Wog_h, const float* __restrict__ b_og,
    const float* __restrict__ colsum, float* __restrict__ out)
{
    extern __shared__ char smem[];
    float*    ecS   = (float*)smem;                   // 16x64  f32
    char*     big   = smem + 4096;                    // 64 KB multi-use region
    _Float16* projS = (_Float16*)big;                 // 16x2048 f16 (stages B-D)
    float*    cueS  = (float*)big;                    // 16x256 f32 (post-D; dead after settle)
    _Float16* stA   = (_Float16*)(big + 16384);
    _Float16* stB   = (_Float16*)(big + 24576);
    float*    ca1F  = (float*)(big + 32768);
    float*    dirF  = (float*)big;                    // reuses cue region post-settle
    _Float16* ca1H  = (_Float16*)(big + 49152);
    _Float16* dirH  = (_Float16*)(big + 57344);
    _Float16* combH = (_Float16*)(big + 16384);       // reuses stA region
    float*    stats = (float*)(smem + 69632);
    float* rowsum = stats;        float* rowsq = stats + 16;
    float* muA    = stats + 32;   float* rsA   = stats + 48;
    float* thr    = stats + 64;   float* ssA   = stats + 80;
    float* ddA    = stats + 96;   float* sdA   = stats + 112;
    float* gateA  = stats + 128;  // 144 floats total

    const int tid  = threadIdx.x;
    const int lane = tid & 31;
    const int wave = tid >> 5;
    const int row0 = blockIdx.x * 16;

    if (tid < 144) stats[tid] = 0.0f;

    // ---- stage A: ec = layernorm(coords + 0.05*mean(coords)) --------------
    {
        int m = tid >> 4;
        int q = tid & 15;
        float x[4]; float s = 0.f, sq = 0.f;
#pragma unroll
        for (int j = 0; j < 4; ++j) {
            int c = q * 4 + j;
            x[j] = coords[(size_t)(row0 + m) * COORD + c]
                 + 0.05f * colsum[c] * (1.0f / (float)B_SZ);
            s += x[j]; sq += x[j] * x[j];
        }
        s  = halfWaveSum(s);
        sq = halfWaveSum(sq);
        float mu = s * (1.0f / COORD);
        float var = sq * (1.0f / COORD) - mu * mu;
        float rs = rsqrtf(var + LN_EPS);
#pragma unroll
        for (int j = 0; j < 4; ++j) {
            int c = q * 4 + j;
            ecS[m * COORD + c] = (x[j] - mu) * rs * ec_gamma[c] + ec_beta[c];
        }
    }
    __syncthreads();

    // ---- stage B: proj = relu(ec @ W_pp^T + b_pp), WMMA 16x2048x64 --------
    {
        half16 a0 = ldsA_f(ecS, COORD, 0, lane);
        half16 a1 = ldsA_f(ecS, COORD, 32, lane);
        float lsum[8], lsq[8];
#pragma unroll
        for (int j = 0; j < 8; ++j) { lsum[j] = 0.f; lsq[j] = 0.f; }
        for (int nt = wave; nt < DG / 16; nt += 8) {
            int nb = nt * 16;
            int n  = nb + (lane & 15);
            float8 c = {};
            c = wmma_f16(a0, gB_h(Wpp_h, COORD, nb, 0, lane), c);
            c = wmma_f16(a1, gB_h(Wpp_h, COORD, nb, 32, lane), c);
            float bias = b_pp[n];
#pragma unroll
            for (int j = 0; j < 8; ++j) {
                float v = fmaxf(c[j] + bias, 0.0f);
                int m = j + ((lane & 16) ? 8 : 0);
                projS[m * DG + n] = (_Float16)v;
                lsum[j] += v; lsq[j] += v * v;
            }
        }
#pragma unroll
        for (int j = 0; j < 8; ++j) {
            float s  = halfWaveSum(lsum[j]);
            float sq = halfWaveSum(lsq[j]);
            if ((lane & 15) == 0) {
                int m = j + ((lane & 16) ? 8 : 0);
                atomicAdd(&rowsum[m], s);
                atomicAdd(&rowsq[m], sq);
            }
        }
    }
    __syncthreads();

    if (tid < 16) {
        float mu  = rowsum[tid] * (1.0f / DG);
        float var = rowsq[tid] * (1.0f / DG) - mu * mu;
        muA[tid] = mu;
        rsA[tid] = rsqrtf(var + LN_EPS);
    }
    __syncthreads();

    // normalize proj tile in place (f16)
    for (int e = tid; e < 16 * DG; e += 256) {
        int m = e >> 11, n = e & (DG - 1);
        float v = ((float)projS[e] - muA[m]) * rsA[m] * dg_gamma[n] + dg_beta[n];
        projS[e] = (_Float16)v;
    }
    __syncthreads();

    // ---- stage C: per-row top-81 threshold by bisection (2 rows per wave) --
#pragma unroll
    for (int rr = 0; rr < 2; ++rr) {
        int m = wave * 2 + rr;
        float vmax = -1e30f, vmin = 1e30f;
        for (int n = lane; n < DG; n += 32) {
            float v = (float)projS[m * DG + n];
            vmax = fmaxf(vmax, v); vmin = fminf(vmin, v);
        }
        float hi = waveMax(vmax) + 1.0f;
        float lo = waveMin(vmin) - 1.0f;
        for (int it = 0; it < 32; ++it) {
            float mid = 0.5f * (lo + hi);
            float cnt = 0.f;
            for (int n = lane; n < DG; n += 32)
                cnt += ((float)projS[m * DG + n] > mid) ? 1.0f : 0.0f;
            cnt = waveSum(cnt);
            if (cnt >= (float)K_ACT) lo = mid; else hi = mid;
        }
        if (lane == 0) thr[m] = lo;
    }
    __syncthreads();

    // ---- stage D: cue = dg_sparse @ W_mossy^T via transposed, coalesced rows
    {
        float acc[2][8];
#pragma unroll
        for (int rr = 0; rr < 2; ++rr)
#pragma unroll
            for (int i = 0; i < 8; ++i) acc[rr][i] = 0.f;
#pragma unroll
        for (int rr = 0; rr < 2; ++rr) {
            int m = wave * 2 + rr;
            float tm = thr[m];
            for (int n = 0; n < DG; ++n) {
                float v = (float)projS[m * DG + n];
                if (v > tm) {
                    const float* wrow = W_mossyT + (size_t)n * CA3 + lane * 8;
#pragma unroll
                    for (int i = 0; i < 8; ++i)
                        acc[rr][i] += v * wrow[i];   // b128-coalesced across lanes
                }
            }
        }
        __syncthreads();  // all proj reads done; safe to overwrite big region
#pragma unroll
        for (int rr = 0; rr < 2; ++rr) {
            int m = wave * 2 + rr;
#pragma unroll
            for (int i = 0; i < 8; ++i) {
                int c = lane * 8 + i;
                cueS[m * CA3 + c] = acc[rr][i];
                stA[m * CA3 + c]  = (_Float16)acc[rr][i];
            }
        }
    }
    __syncthreads();

    // ---- stage E: 5 settle steps, WMMA 16x256x256 each --------------------
    {
        _Float16* cur = stA;
        _Float16* nxt = stB;
        for (int it = 0; it < STEPS; ++it) {
            for (int nt = wave; nt < 16; nt += 8) {
                int nb = nt * 16;
                int n  = nb + (lane & 15);
                float8 c = {};
                for (int kk = 0; kk < CA3; kk += 32)
                    c = wmma_f16(ldsA_h(cur, CA3, kk, lane),
                                 gB_h(Wrec_h, CA3, nb, kk, lane), c);
#pragma unroll
                for (int j = 0; j < 8; ++j) {
                    int m = j + ((lane & 16) ? 8 : 0);
                    float s = tanhf(GAIN * c[j] + (1.0f - GAIN) * cueS[m * CA3 + n]);
                    nxt[m * CA3 + n] = (_Float16)s;
                }
            }
            __syncthreads();
            _Float16* t = cur; cur = nxt; nxt = t;
        }
        // final state in stB (== cur)

        // ---- stage F: ca1_input and direct --------------------------------
        for (int nt = wave; nt < 16; nt += 8) {
            int nb = nt * 16;
            int n  = nb + (lane & 15);
            float8 c = {};
            for (int kk = 0; kk < CA3; kk += 32)
                c = wmma_f16(ldsA_h(cur, CA3, kk, lane),
                             gB_h(Wsc_h, CA3, nb, kk, lane), c);
            float bias = b_sc[n];
#pragma unroll
            for (int j = 0; j < 8; ++j) {
                int m = j + ((lane & 16) ? 8 : 0);
                float v = c[j] + bias;
                ca1F[m * CA1 + n] = v;
                ca1H[m * CA1 + n] = (_Float16)v;
            }
        }
        {
            half16 e0 = ldsA_f(ecS, COORD, 0, lane);
            half16 e1 = ldsA_f(ecS, COORD, 32, lane);
            for (int nt = wave; nt < 16; nt += 8) {
                int nb = nt * 16;
                int n  = nb + (lane & 15);
                float8 c = {};
                c = wmma_f16(e0, gB_h(Wta_h, COORD, nb, 0, lane), c);
                c = wmma_f16(e1, gB_h(Wta_h, COORD, nb, 32, lane), c);
                float bias = b_ta[n];
#pragma unroll
                for (int j = 0; j < 8; ++j) {
                    int m = j + ((lane & 16) ? 8 : 0);
                    float v = c[j] + bias;
                    dirF[m * CA1 + n] = v;
                    dirH[m * CA1 + n] = (_Float16)v;
                }
            }
        }
    }
    __syncthreads();

    // ---- stage G: cosine gate from s_proj / d_proj (never materialized) ----
    {
        float ssL[8], ddL[8], sdL[8];
#pragma unroll
        for (int j = 0; j < 8; ++j) { ssL[j] = 0.f; ddL[j] = 0.f; sdL[j] = 0.f; }
        for (int nt = wave; nt < 16; nt += 8) {
            int nb = nt * 16;
            float8 cs = {}; float8 cd = {};
            for (int kk = 0; kk < CA1; kk += 32) {
                cs = wmma_f16(ldsA_h(ca1H, CA1, kk, lane),
                              gB_h(Wcs_h, CA1, nb, kk, lane), cs);
                cd = wmma_f16(ldsA_h(dirH, CA1, kk, lane),
                              gB_h(Wcd_h, CA1, nb, kk, lane), cd);
            }
#pragma unroll
            for (int j = 0; j < 8; ++j) {
                float s = cs[j], d = cd[j];
                ssL[j] += s * s; ddL[j] += d * d; sdL[j] += s * d;
            }
        }
#pragma unroll
        for (int j = 0; j < 8; ++j) {
            float s = halfWaveSum(ssL[j]);
            float d = halfWaveSum(ddL[j]);
            float x = halfWaveSum(sdL[j]);
            if ((lane & 15) == 0) {
                int m = j + ((lane & 16) ? 8 : 0);
                atomicAdd(&ssA[m], s);
                atomicAdd(&ddA[m], d);
                atomicAdd(&sdA[m], x);
            }
        }
    }
    __syncthreads();
    if (tid < 16) {
        float sn = fmaxf(sqrtf(ssA[tid]), COS_EPSF);
        float dn = fmaxf(sqrtf(ddA[tid]), COS_EPSF);
        float cosv = sdA[tid] / (sn * dn);
        float nov  = fminf(fmaxf(1.0f - cosv, 0.0f), 1.0f);
        gateA[tid] = nov;
        out[(size_t)B_SZ * CA1 + row0 + tid] = nov;   // novelty output
    }
    __syncthreads();

    // ---- stage H: combined, output GEMM + tanh -----------------------------
    for (int e = tid; e < 16 * CA1; e += 256) {
        int m = e >> 8;
        float g = gateA[m];
        combH[e] = (_Float16)(g * dirF[e] + (1.0f - g) * ca1F[e]);
    }
    __syncthreads();
    for (int nt = wave; nt < 16; nt += 8) {
        int nb = nt * 16;
        int n  = nb + (lane & 15);
        float8 c = {};
        for (int kk = 0; kk < CA1; kk += 32)
            c = wmma_f16(ldsA_h(combH, CA1, kk, lane),
                         gB_h(Wog_h, CA1, nb, kk, lane), c);
        float bias = b_og[n];
#pragma unroll
        for (int j = 0; j < 8; ++j) {
            int m = j + ((lane & 16) ? 8 : 0);
            out[(size_t)(row0 + m) * CA1 + n] = tanhf(c[j] + bias);
        }
    }
}

extern "C" void kernel_launch(void* const* d_in, const int* in_sizes, int n_in,
                              void* d_out, int out_size, void* d_ws, size_t ws_size,
                              hipStream_t stream) {
    const float* coords   = (const float*)d_in[0];
    const float* ec_gamma = (const float*)d_in[1];
    const float* ec_beta  = (const float*)d_in[2];
    const float* W_pp     = (const float*)d_in[3];
    const float* b_pp     = (const float*)d_in[4];
    const float* dg_gamma = (const float*)d_in[5];
    const float* dg_beta  = (const float*)d_in[6];
    const float* W_mossy  = (const float*)d_in[7];
    const float* W_rec    = (const float*)d_in[8];
    const float* W_sc     = (const float*)d_in[9];
    const float* b_sc     = (const float*)d_in[10];
    const float* W_ta     = (const float*)d_in[11];
    const float* b_ta     = (const float*)d_in[12];
    const float* W_cs     = (const float*)d_in[13];
    const float* W_cd     = (const float*)d_in[14];
    const float* W_og     = (const float*)d_in[15];
    const float* b_og     = (const float*)d_in[16];

    // -------- workspace layout --------
    char* ws = (char*)d_ws;
    float*    colsum   = (float*)ws;                         // 64 f32 (pad to 1 KB)
    float*    W_mossyT = (float*)(ws + 1024);                // 2048*256 f32 = 2 MB
    char*     h        = ws + 1024 + (size_t)DG * CA3 * 4;
    _Float16* Wpp_h = (_Float16*)h;                          h += (size_t)DG  * COORD * 2;
    _Float16* Wrec_h = (_Float16*)h;                         h += (size_t)CA3 * CA3   * 2;
    _Float16* Wsc_h  = (_Float16*)h;                         h += (size_t)CA1 * CA3   * 2;
    _Float16* Wta_h  = (_Float16*)h;                         h += (size_t)CA1 * COORD * 2;
    _Float16* Wcs_h  = (_Float16*)h;                         h += (size_t)CA1 * CA1   * 2;
    _Float16* Wcd_h  = (_Float16*)h;                         h += (size_t)CA1 * CA1   * 2;
    _Float16* Wog_h  = (_Float16*)h;                         h += (size_t)CA1 * CA1   * 2;

    float* out = (float*)d_out;

    const size_t SMEM_BYTES = 69632 + 144 * sizeof(float);   // ~70.2 KB (CDNA5: 320 KB/WGP)
    hipFuncSetAttribute((const void*)hippo_fused,
                        hipFuncAttributeMaxDynamicSharedMemorySize, (int)SMEM_BYTES);

    hipMemsetAsync(d_ws, 0, COORD * sizeof(float), stream);
    hippo_colsum<<<B_SZ / 128, 256, 0, stream>>>(coords, colsum);

    mossy_T<<<(DG * CA3) / 256, 256, 0, stream>>>(W_mossy, W_mossyT);
    cvt_f16<<<(DG  * COORD) / 256, 256, 0, stream>>>(W_pp,  Wpp_h,  DG  * COORD);
    cvt_f16<<<(CA3 * CA3)   / 256, 256, 0, stream>>>(W_rec, Wrec_h, CA3 * CA3);
    cvt_f16<<<(CA1 * CA3)   / 256, 256, 0, stream>>>(W_sc,  Wsc_h,  CA1 * CA3);
    cvt_f16<<<(CA1 * COORD) / 256, 256, 0, stream>>>(W_ta,  Wta_h,  CA1 * COORD);
    cvt_f16<<<(CA1 * CA1)   / 256, 256, 0, stream>>>(W_cs,  Wcs_h,  CA1 * CA1);
    cvt_f16<<<(CA1 * CA1)   / 256, 256, 0, stream>>>(W_cd,  Wcd_h,  CA1 * CA1);
    cvt_f16<<<(CA1 * CA1)   / 256, 256, 0, stream>>>(W_og,  Wog_h,  CA1 * CA1);

    hippo_fused<<<B_SZ / 16, 256, SMEM_BYTES, stream>>>(
        coords, ec_gamma, ec_beta, Wpp_h, b_pp, dg_gamma, dg_beta, W_mossyT,
        Wrec_h, Wsc_h, b_sc, Wta_h, b_ta, Wcs_h, Wcd_h, Wog_h, b_og, colsum, out);
}
